// MemNetBlockCell_32040456028436
// MI455X (gfx1250) — matrix operand order repaired and checked by reading, not compiled, not run on hardware
//
#include <hip/hip_runtime.h>
#include <hip/hip_bf16.h>

typedef _Float16 v16h __attribute__((ext_vector_type(16)));
typedef _Float16 v8h  __attribute__((ext_vector_type(8)));
typedef _Float16 v4h  __attribute__((ext_vector_type(4)));
typedef float    v8f  __attribute__((ext_vector_type(8)));

union HFrag { v16h f; v8h h[2]; };

#define Bn 512
#define Ln 128
#define Dn 512
#define Mn (Bn*Ln)          // 65536
#define PIT 40              // LDS pitch (halves) for 32-wide K tiles
#define PITA 136            // LDS pitch (halves) for 128-wide K tiles

// ---------------------------------------------------------------------------
// K0: u = Wx^T (Wq^T w2),  consts[0] = bx·v + bq·w2   (s_q = x·u + consts[0])
// ---------------------------------------------------------------------------
__global__ __launch_bounds__(512) void k_precompute(
    const float* __restrict__ Wx, const float* __restrict__ bx,
    const float* __restrict__ Wq, const float* __restrict__ bq,
    const float* __restrict__ w_mlp, float* __restrict__ uA,
    float* __restrict__ consts) {
  __shared__ float vs[Dn];
  __shared__ float red[Dn];
  const int t = threadIdx.x;
  float v = 0.f;
  for (int dp = 0; dp < Dn; ++dp) v += w_mlp[Dn + dp] * Wq[(size_t)dp*Dn + t];
  vs[t] = v;
  __syncthreads();
  float uu = 0.f;
  for (int d = 0; d < Dn; ++d) uu += vs[d] * Wx[(size_t)d*Dn + t];
  uA[t] = uu;
  red[t] = bx[t]*vs[t] + bq[t]*w_mlp[Dn + t];
  __syncthreads();
  for (int s = 256; s > 0; s >>= 1) {
    if (t < s) red[t] += red[t+s];
    __syncthreads();
  }
  if (t == 0) consts[0] = red[0];
}

// ---------------------------------------------------------------------------
// K0b: one-time f32 -> f16 conversion of Wk and Wp
// ---------------------------------------------------------------------------
__global__ __launch_bounds__(256) void k_cvt_w(
    const float* __restrict__ Wk, const float* __restrict__ Wp,
    _Float16* __restrict__ Wkh, _Float16* __restrict__ Wph) {
  size_t i = ((size_t)blockIdx.x * 256 + threadIdx.x) * 4;
  const float4 a = *reinterpret_cast<const float4*>(Wk + i);
  v4h ha = { (_Float16)a.x, (_Float16)a.y, (_Float16)a.z, (_Float16)a.w };
  *reinterpret_cast<v4h*>(Wkh + i) = ha;
  const float4 p = *reinterpret_cast<const float4*>(Wp + i);
  v4h hp = { (_Float16)p.x, (_Float16)p.y, (_Float16)p.z, (_Float16)p.w };
  *reinterpret_cast<v4h*>(Wph + i) = hp;
}

// ---------------------------------------------------------------------------
// K1: gamma = y@Wg^T + bg + 1 ; beta = y@Wb^T + bb   (per batch block)
// ---------------------------------------------------------------------------
__global__ __launch_bounds__(256) void k_gamma_beta(
    const float* __restrict__ y,
    const float* __restrict__ Wg, const float* __restrict__ bg,
    const float* __restrict__ Wb, const float* __restrict__ bb,
    float* __restrict__ gammaA, float* __restrict__ betaA) {
  __shared__ float ys[Dn];
  const int b = blockIdx.x, t = threadIdx.x;
  ys[t]       = y[(size_t)b*Dn + t];
  ys[t + 256] = y[(size_t)b*Dn + 256 + t];
  __syncthreads();
  const int w = t >> 5, lane = t & 31;
  for (int l = w; l < Ln; l += 8) {
    float ag = 0.f, ab = 0.f;
    #pragma unroll
    for (int j = 0; j < 16; ++j) {
      int d = lane + 32*j;
      float yy = ys[d];
      ag += yy * Wg[(size_t)l*Dn + d];
      ab += yy * Wb[(size_t)l*Dn + d];
    }
    #pragma unroll
    for (int o = 16; o; o >>= 1) {
      ag += __shfl_xor(ag, o, 32);
      ab += __shfl_xor(ab, o, 32);
    }
    if (lane == 0) {
      gammaA[(size_t)b*Ln + l] = ag + bg[l] + 1.0f;
      betaA [(size_t)b*Ln + l] = ab + bb[l];
    }
  }
}

// ---------------------------------------------------------------------------
// K2: kx = x @ Wk^T + bk  (f16 out), fused s_q = x·u + c, s_k = kx·w1
//     Block: 32 rows × full N=512, 8 waves (2x4), register-pipelined staging
// ---------------------------------------------------------------------------
__global__ __launch_bounds__(256) void k_gemm_kx(
    const float* __restrict__ x, const _Float16* __restrict__ Wkh,
    const float* __restrict__ bk, const float* __restrict__ uA,
    const float* __restrict__ consts, const float* __restrict__ w_mlp,
    _Float16* __restrict__ kxh, float* __restrict__ s_q,
    float* __restrict__ s_k) {
  __shared__ _Float16 As[32*PIT];
  __shared__ _Float16 Bs[Dn*PIT];
  __shared__ float sk_s[32];
  const int t = threadIdx.x;
  const int m0 = blockIdx.x * 32;
  const int w = t >> 5, lane = t & 31, g = lane >> 4, col = lane & 15;
  const int wm = w & 1, wn = w >> 1;
  const int rowA = t >> 3, cgA = (t & 7) * 4;

  v8f acc[8] = {};
  float sq = 0.f;
  if (t < 32) sk_s[t] = 0.f;

  // register-staged tile for kk=0
  const float* xRow = x + (size_t)(m0 + rowA)*Dn + cgA;
  float4 aReg = *reinterpret_cast<const float4*>(xRow);
  v8h bReg[8];
  #pragma unroll
  for (int it = 0; it < 8; ++it) {
    int idx = t + it*256;                 // 0..2047
    int row = idx >> 2, cg = (idx & 3) * 8;
    bReg[it] = *reinterpret_cast<const v8h*>(Wkh + (size_t)row*Dn + cg);
  }

  for (int kk = 0; kk < Dn; kk += 32) {
    // commit staged registers to LDS
    {
      v4h hv = { (_Float16)aReg.x, (_Float16)aReg.y, (_Float16)aReg.z, (_Float16)aReg.w };
      *reinterpret_cast<v4h*>(&As[rowA*PIT + cgA]) = hv;
    }
    #pragma unroll
    for (int it = 0; it < 8; ++it) {
      int idx = t + it*256;
      int row = idx >> 2, cg = (idx & 3) * 8;
      *reinterpret_cast<v8h*>(&Bs[row*PIT + cg]) = bReg[it];
    }
    __syncthreads();

    // issue next tile's global loads (overlap with WMMA below)
    if (kk + 32 < Dn) {
      aReg = *reinterpret_cast<const float4*>(xRow + kk + 32);
      #pragma unroll
      for (int it = 0; it < 8; ++it) {
        int idx = t + it*256;
        int row = idx >> 2, cg = (idx & 3) * 8;
        bReg[it] = *reinterpret_cast<const v8h*>(Wkh + (size_t)row*Dn + kk + 32 + cg);
      }
    }

    // fused s_q partial (x·u) from staged A tile
    if (t < 32) {
      #pragma unroll
      for (int k = 0; k < 32; ++k) sq += (float)As[t*PIT + k] * uA[kk + k];
    }

    // WMMA: batch fragment loads (4 tiles) then issue the chain
    HFrag a;
    const int arow = wm*16 + col;
    a.h[0] = *reinterpret_cast<const v8h*>(&As[arow*PIT + 8*g]);
    a.h[1] = *reinterpret_cast<const v8h*>(&As[arow*PIT + 16 + 8*g]);
    #pragma unroll
    for (int jb = 0; jb < 2; ++jb) {
      HFrag bf[4];
      #pragma unroll
      for (int j2 = 0; j2 < 4; ++j2) {
        int n = wn*128 + (jb*4 + j2)*16 + col;
        bf[j2].h[0] = *reinterpret_cast<const v8h*>(&Bs[n*PIT + 8*g]);
        bf[j2].h[1] = *reinterpret_cast<const v8h*>(&Bs[n*PIT + 16 + 8*g]);
      }
      #pragma unroll
      for (int j2 = 0; j2 < 4; ++j2)
        acc[jb*4 + j2] = __builtin_amdgcn_wmma_f32_16x16x32_f16(
            false, a.f, false, bf[j2].f, (short)0, acc[jb*4 + j2], false, false);
    }
    __syncthreads();
  }

  // epilogue: bias, f16 store, fused s_k = kx·w1
  float skp[8] = {};
  #pragma unroll
  for (int j = 0; j < 8; ++j) {
    int n = wn*128 + j*16 + col;
    float bkn = bk[n], w1n = w_mlp[n];
    #pragma unroll
    for (int i = 0; i < 8; ++i) {
      int m = m0 + wm*16 + 8*g + i;
      float val = acc[j][i] + bkn;
      kxh[(size_t)m*Dn + n] = (_Float16)val;
      skp[i] += val * w1n;
    }
  }
  #pragma unroll
  for (int i = 0; i < 8; ++i) atomicAdd(&sk_s[wm*16 + 8*g + i], skp[i]);
  if (t < 32) s_q[m0 + t] = sq + consts[0];
  __syncthreads();
  if (t < 32) s_k[m0 + t] = sk_s[t];
}

// ---------------------------------------------------------------------------
// K3: per-batch attention: score=tanh(s_q+s_k), softmax, ctx = attn @ kx
//     ctx overwrites kx in place (each chunk staged to LDS before store)
// ---------------------------------------------------------------------------
__global__ __launch_bounds__(256) void k_attention(
    _Float16* __restrict__ kc, const float* __restrict__ s_q,
    const float* __restrict__ s_k) {
  __shared__ _Float16 attn_s[Ln*PITA];
  __shared__ _Float16 Bs[Ln*PITA];
  __shared__ float sqs[Ln], sks[Ln];
  const int b = blockIdx.x, t = threadIdx.x;
  const int w = t >> 5, lane = t & 31, g = lane >> 4, col = lane & 15;

  if (t < Ln) { sqs[t] = s_q[(size_t)b*Ln + t]; sks[t] = s_k[(size_t)b*Ln + t]; }
  __syncthreads();

  // score + softmax: each wave owns 16 rows
  for (int q = w; q < Ln; q += 8) {
    float sqv = sqs[q];
    float sc[4];
    float mx = -3.0e38f;
    #pragma unroll
    for (int r = 0; r < 4; ++r) {
      sc[r] = tanhf(sqv + sks[lane + 32*r]);
      mx = fmaxf(mx, sc[r]);
    }
    #pragma unroll
    for (int o = 16; o; o >>= 1) mx = fmaxf(mx, __shfl_xor(mx, o, 32));
    float sum = 0.f;
    #pragma unroll
    for (int r = 0; r < 4; ++r) { sc[r] = __expf(sc[r] - mx); sum += sc[r]; }
    #pragma unroll
    for (int o = 16; o; o >>= 1) sum += __shfl_xor(sum, o, 32);
    float inv = 1.0f / sum;
    #pragma unroll
    for (int r = 0; r < 4; ++r)
      attn_s[q*PITA + lane + 32*r] = (_Float16)(sc[r]*inv);
  }

  const int wm = w >> 2, wn = w & 3;   // 2 (M) x 4 (N) wave grid
  const int n0 = (t & 15) * 8;
  for (int nc = 0; nc < 4; ++nc) {
    __syncthreads();
    // stage kx chunk [k][n-chunk] transposed into Bs[n][k]; batch global loads
    #pragma unroll
    for (int ib = 0; ib < 2; ++ib) {
      v8h vv[4];
      #pragma unroll
      for (int it2 = 0; it2 < 4; ++it2) {
        int k = (t >> 4) + (ib*4 + it2)*16;
        vv[it2] = *reinterpret_cast<const v8h*>(kc + ((size_t)b*Ln + k)*Dn + nc*128 + n0);
      }
      #pragma unroll
      for (int it2 = 0; it2 < 4; ++it2) {
        int k = (t >> 4) + (ib*4 + it2)*16;
        #pragma unroll
        for (int i = 0; i < 8; ++i) Bs[(n0+i)*PITA + k] = vv[it2][i];
      }
    }
    __syncthreads();

    v8f acc[4][2] = {};
    #pragma unroll
    for (int ks = 0; ks < 4; ++ks) {
      int k0 = ks*32;
      HFrag a[4];
      #pragma unroll
      for (int mt = 0; mt < 4; ++mt) {
        int arow = wm*64 + mt*16 + col;
        a[mt].h[0] = *reinterpret_cast<const v8h*>(&attn_s[arow*PITA + k0 + 8*g]);
        a[mt].h[1] = *reinterpret_cast<const v8h*>(&attn_s[arow*PITA + k0 + 16 + 8*g]);
      }
      #pragma unroll
      for (int nt = 0; nt < 2; ++nt) {
        int n = wn*32 + nt*16 + col;
        HFrag bf;
        bf.h[0] = *reinterpret_cast<const v8h*>(&Bs[n*PITA + k0 + 8*g]);
        bf.h[1] = *reinterpret_cast<const v8h*>(&Bs[n*PITA + k0 + 16 + 8*g]);
        #pragma unroll
        for (int mt = 0; mt < 4; ++mt)
          acc[mt][nt] = __builtin_amdgcn_wmma_f32_16x16x32_f16(
              false, a[mt].f, false, bf.f, (short)0, acc[mt][nt], false, false);
      }
    }
    // store ctx chunk (overwrites kx region already consumed into LDS)
    #pragma unroll
    for (int mt = 0; mt < 4; ++mt)
      #pragma unroll
      for (int nt = 0; nt < 2; ++nt)
        #pragma unroll
        for (int i = 0; i < 8; ++i) {
          int m = wm*64 + mt*16 + 8*g + i;
          int n = wn*32 + nt*16 + col;
          kc[((size_t)b*Ln + m)*Dn + nc*128 + n] = (_Float16)acc[mt][nt][i];
        }
  }
}

// ---------------------------------------------------------------------------
// K4: out = relu(gamma*(ctx@Wp^T + bp) + beta) + x
// ---------------------------------------------------------------------------
__global__ __launch_bounds__(256) void k_gemm_out(
    const _Float16* __restrict__ ctx, const _Float16* __restrict__ Wph,
    const float* __restrict__ bp, const float* __restrict__ gammaA,
    const float* __restrict__ betaA, const float* __restrict__ x,
    float* __restrict__ out) {
  __shared__ _Float16 As[32*PIT];
  __shared__ _Float16 Bs[Dn*PIT];
  const int t = threadIdx.x;
  const int m0 = blockIdx.x * 32;
  const int w = t >> 5, lane = t & 31, g = lane >> 4, col = lane & 15;
  const int wm = w & 1, wn = w >> 1;
  const int rowA = t >> 3, cgA = (t & 7) * 4;

  v8f acc[8] = {};

  const _Float16* ctxRow = ctx + (size_t)(m0 + rowA)*Dn + cgA;
  v4h aReg = *reinterpret_cast<const v4h*>(ctxRow);
  v8h bReg[8];
  #pragma unroll
  for (int it = 0; it < 8; ++it) {
    int idx = t + it*256;
    int row = idx >> 2, cg = (idx & 3) * 8;
    bReg[it] = *reinterpret_cast<const v8h*>(Wph + (size_t)row*Dn + cg);
  }

  for (int kk = 0; kk < Dn; kk += 32) {
    *reinterpret_cast<v4h*>(&As[rowA*PIT + cgA]) = aReg;
    #pragma unroll
    for (int it = 0; it < 8; ++it) {
      int idx = t + it*256;
      int row = idx >> 2, cg = (idx & 3) * 8;
      *reinterpret_cast<v8h*>(&Bs[row*PIT + cg]) = bReg[it];
    }
    __syncthreads();

    if (kk + 32 < Dn) {
      aReg = *reinterpret_cast<const v4h*>(ctxRow + kk + 32);
      #pragma unroll
      for (int it = 0; it < 8; ++it) {
        int idx = t + it*256;
        int row = idx >> 2, cg = (idx & 3) * 8;
        bReg[it] = *reinterpret_cast<const v8h*>(Wph + (size_t)row*Dn + kk + 32 + cg);
      }
    }

    HFrag a;
    const int arow = wm*16 + col;
    a.h[0] = *reinterpret_cast<const v8h*>(&As[arow*PIT + 8*g]);
    a.h[1] = *reinterpret_cast<const v8h*>(&As[arow*PIT + 16 + 8*g]);
    #pragma unroll
    for (int jb = 0; jb < 2; ++jb) {
      HFrag bf[4];
      #pragma unroll
      for (int j2 = 0; j2 < 4; ++j2) {
        int n = wn*128 + (jb*4 + j2)*16 + col;
        bf[j2].h[0] = *reinterpret_cast<const v8h*>(&Bs[n*PIT + 8*g]);
        bf[j2].h[1] = *reinterpret_cast<const v8h*>(&Bs[n*PIT + 16 + 8*g]);
      }
      #pragma unroll
      for (int j2 = 0; j2 < 4; ++j2)
        acc[jb*4 + j2] = __builtin_amdgcn_wmma_f32_16x16x32_f16(
            false, a.f, false, bf[j2].f, (short)0, acc[jb*4 + j2], false, false);
    }
    __syncthreads();
  }

  float gm[8], bt[8];
  #pragma unroll
  for (int i = 0; i < 8; ++i) {
    int m = m0 + wm*16 + 8*g + i;
    gm[i] = gammaA[m];
    bt[i] = betaA[m];
  }
  #pragma unroll
  for (int j = 0; j < 8; ++j) {
    int n = wn*128 + j*16 + col;
    float bpn = bp[n];
    #pragma unroll
    for (int i = 0; i < 8; ++i) {
      int m = m0 + wm*16 + 8*g + i;
      float val = acc[j][i] + bpn;
      val = gm[i]*val + bt[i];
      float xv = __builtin_nontemporal_load(x + (size_t)m*Dn + n);
      val = fmaxf(val, 0.0f) + xv;
      __builtin_nontemporal_store(val, out + (size_t)m*Dn + n);
    }
  }
}

// ---------------------------------------------------------------------------
extern "C" void kernel_launch(void* const* d_in, const int* in_sizes, int n_in,
                              void* d_out, int out_size, void* d_ws, size_t ws_size,
                              hipStream_t stream) {
  const float* x     = (const float*)d_in[0];
  const float* y     = (const float*)d_in[1];
  const float* Wg    = (const float*)d_in[2];
  const float* bg    = (const float*)d_in[3];
  const float* Wb    = (const float*)d_in[4];
  const float* bb    = (const float*)d_in[5];
  const float* Wx    = (const float*)d_in[6];
  const float* bx    = (const float*)d_in[7];
  const float* Wk    = (const float*)d_in[8];
  const float* bk    = (const float*)d_in[9];
  const float* Wq    = (const float*)d_in[10];
  const float* bq    = (const float*)d_in[11];
  const float* Wp    = (const float*)d_in[12];
  const float* bp    = (const float*)d_in[13];
  const float* w_mlp = (const float*)d_in[14];

  char* ws = (char*)d_ws;
  size_t off = 0;
  _Float16* kc  = (_Float16*)(ws + off); off += (size_t)Mn*Dn*2;   // kx, then ctx
  float* sqA    = (float*)(ws + off); off += (size_t)Mn*4;
  float* skA    = (float*)(ws + off); off += (size_t)Mn*4;
  float* gammaA = (float*)(ws + off); off += (size_t)Mn*4;
  float* betaA  = (float*)(ws + off); off += (size_t)Mn*4;
  float* uA     = (float*)(ws + off); off += (size_t)Dn*4;
  float* consts = (float*)(ws + off); off += 256;
  _Float16* Wkh = (_Float16*)(ws + off); off += (size_t)Dn*Dn*2;
  _Float16* Wph = (_Float16*)(ws + off); off += (size_t)Dn*Dn*2;
  float* out = (float*)d_out;

  k_precompute<<<1, 512, 0, stream>>>(Wx, bx, Wq, bq, w_mlp, uA, consts);
  k_cvt_w<<<(Dn*Dn/4)/256, 256, 0, stream>>>(Wk, Wp, Wkh, Wph);
  k_gamma_beta<<<Bn, 256, 0, stream>>>(y, Wg, bg, Wb, bb, gammaA, betaA);
  k_gemm_kx<<<Mn/32, 256, 0, stream>>>(x, Wkh, bk, uA, consts, w_mlp, kc, sqA, skA);
  k_attention<<<Bn, 256, 0, stream>>>(kc, sqA, skA);
  k_gemm_out<<<Mn/32, 256, 0, stream>>>(kc, Wph, bp, gammaA, betaA, x, out);
}
